// Net_85581518340249
// MI455X (gfx1250) — compile-verified, matrix operand order checked by reading, not run on hardware
//
#include <hip/hip_runtime.h>
#include <hip/hip_bf16.h>
#include <math.h>

// ---------------------------------------------------------------------------
// Types for WMMA operands (wave32, CDNA5): 16x16x32 f16 -> f32
// ---------------------------------------------------------------------------
typedef _Float16 v16h __attribute__((ext_vector_type(16)));
typedef _Float16 v8h  __attribute__((ext_vector_type(8)));
typedef float    v8f  __attribute__((ext_vector_type(8)));

#define NNODES 50000
#define NEDGES 400000
#define MAXCHUNKS 6           // K <= 192

// ---------------------------------------------------------------------------
// Pack a weight matrix W[fin x 32] (f32, row-major) into the CDNA5 16-bit
// B-operand layout: out[((chunk*2 + ntile)*32 + lane)*16 + e] with
//   k   = chunk*32 + e + 8*(lane>>4) + (e>=8 ? 8 : 0)
//   col = ntile*16 + (lane & 15)
// so the GEMM wave reads one contiguous 32B v16h per lane.
// grid = chunks*2 blocks of 32 threads.
// ---------------------------------------------------------------------------
__global__ void pack_w_kernel(const float* __restrict__ W, _Float16* __restrict__ out,
                              int fin) {
  int l   = threadIdx.x;
  int c   = blockIdx.x >> 1;
  int t   = blockIdx.x & 1;
  int hl  = l >> 4;
  int col = t * 16 + (l & 15);
#pragma unroll
  for (int e = 0; e < 16; ++e) {
    int k = c * 32 + e + 8 * hl + (e >= 8 ? 8 : 0);
    float v = (k < fin) ? W[(size_t)k * 32 + col] : 0.0f;
    out[((size_t)(c * 2 + t) * 32 + l) * 16 + e] = (_Float16)v;
  }
}

// ---------------------------------------------------------------------------
// Build the concatenated, f16, K-padded A matrix  Ain[E x kpad].
// mode 1: conv1  (kpad=32):  [edge_attr(4) | 0...]
// mode 2: conv2  (kpad=128): [x1[row] | x1[col] | edge_attr | e1]
// mode 3: conv3/4(kpad=192): [nA[row] | nB[row] | nA[col] | nB[col] | eP | eQ]
// ---------------------------------------------------------------------------
__global__ void pack_a_kernel(_Float16* __restrict__ Ain, int kpad, int mode,
                              const int* __restrict__ rowI, const int* __restrict__ colI,
                              const float* __restrict__ ea,
                              const float* __restrict__ nA, const float* __restrict__ nB,
                              const _Float16* __restrict__ eP, const _Float16* __restrict__ eQ,
                              int nEdges) {
  int idx = blockIdx.x * blockDim.x + threadIdx.x;
  long total = (long)nEdges * kpad;
  if (idx >= total) return;
  int e = idx / kpad;
  int k = idx - e * kpad;
  float v = 0.0f;
  if (mode == 1) {
    if (k < 4) v = ea[(size_t)e * 4 + k];
  } else if (mode == 2) {
    if      (k < 32)  v = nA[(size_t)rowI[e] * 32 + k];
    else if (k < 64)  v = nA[(size_t)colI[e] * 32 + (k - 32)];
    else if (k < 68)  v = ea[(size_t)e * 4 + (k - 64)];
    else if (k < 100) v = (float)eP[(size_t)e * 32 + (k - 68)];
  } else {
    if      (k < 32)  v = nA[(size_t)rowI[e] * 32 + k];
    else if (k < 64)  v = nB[(size_t)rowI[e] * 32 + (k - 32)];
    else if (k < 96)  v = nA[(size_t)colI[e] * 32 + (k - 64)];
    else if (k < 128) v = nB[(size_t)colI[e] * 32 + (k - 96)];
    else if (k < 160) v = (float)eP[(size_t)e * 32 + (k - 128)];
    else              v = (float)eQ[(size_t)e * 32 + (k - 160)];
  }
  Ain[idx] = (_Float16)v;
}

// ---------------------------------------------------------------------------
// Core edge-MLP:  Eout = relu(A@W0+b0)@W1+b1 ; scatter pre-relu into node sums.
// One wave per 16-edge tile; 8 waves/block; 25000 tiles total (exact grid).
//
// Weights (already in B-operand layout) are staged into LDS ONCE per block
// with GLOBAL_LOAD_ASYNC_TO_LDS_B128 (ASYNCcnt), so the K-loop's B reads are
// ds_load_b128 that co-issue with the A global_load_b128s, and each WMMA only
// waits on A's LOADcnt. Hidden tile is bounced through LDS for the cross-lane
// transpose into the A-operand layout, then 2 more WMMAs for the 32x32 GEMM.
// ---------------------------------------------------------------------------
__global__ __launch_bounds__(256)
void edge_mlp_wmma_kernel(const _Float16* __restrict__ Ain, int kchunks, int kpad,
                          const _Float16* __restrict__ W0p, const float* __restrict__ bias0,
                          const _Float16* __restrict__ W1p, const float* __restrict__ bias1,
                          _Float16* __restrict__ Eout,
                          const int* __restrict__ rowI, float* __restrict__ sAcc,
                          int doScatter, int nEdges) {
  __shared__ _Float16 wlds[MAXCHUNKS * 1024 + 1024];  // 12KB W0 + 2KB W1
  __shared__ _Float16 hid[8][16 * 32];                // 1 KB per wave

  int wave = threadIdx.x >> 5;
  int lane = threadIdx.x & 31;

  // ---- stage packed weights into LDS (wave 0, async DMA path) ----
  if (wave == 0) {
    unsigned ldsW0 = (unsigned)(unsigned long long)(uintptr_t)&wlds[0];
    unsigned ldsW1 = (unsigned)(unsigned long long)(uintptr_t)&wlds[MAXCHUNKS * 1024];
    int nvec0 = kchunks * 128;          // 16B vectors of W0 operand data
    for (int i = lane; i < nvec0; i += 32) {
      unsigned l = ldsW0 + (unsigned)i * 16u;
      unsigned long long g = (unsigned long long)(uintptr_t)W0p + (unsigned long long)i * 16ull;
      asm volatile("global_load_async_to_lds_b128 %0, %1, off"
                   :: "v"(l), "v"(g) : "memory");
    }
    for (int i = lane; i < 128; i += 32) {
      unsigned l = ldsW1 + (unsigned)i * 16u;
      unsigned long long g = (unsigned long long)(uintptr_t)W1p + (unsigned long long)i * 16ull;
      asm volatile("global_load_async_to_lds_b128 %0, %1, off"
                   :: "v"(l), "v"(g) : "memory");
    }
    asm volatile("s_wait_asynccnt 0x0" ::: "memory");
  }
  __syncthreads();

  int tile = blockIdx.x * 8 + wave;
  int tileBase = tile * 16;
  if (tileBase >= nEdges) return;        // uniform per wave -> EXEC all-1s at WMMAs

  int hl  = lane >> 4;                   // which 16-lane half
  int l15 = lane & 15;

  // ---- GEMM 1: [16 x kpad] @ [kpad x 32], B from LDS ----
  v8f c0 = {}; v8f c1 = {};
  const _Float16* arow = Ain + (size_t)(tileBase + l15) * kpad + 8 * hl;
  for (int c = 0; c < kchunks; ++c) {
    v8h alo = *(const v8h*)(arow + c * 32);
    v8h ahi = *(const v8h*)(arow + c * 32 + 16);
    v16h a = __builtin_shufflevector(alo, ahi, 0,1,2,3,4,5,6,7,8,9,10,11,12,13,14,15);
    v16h b0v = *(const v16h*)(&wlds[((c * 2 + 0) * 32 + lane) * 16]);
    v16h b1v = *(const v16h*)(&wlds[((c * 2 + 1) * 32 + lane) * 16]);
    c0 = __builtin_amdgcn_wmma_f32_16x16x32_f16(false, a, false, b0v, (short)0, c0, false, false);
    c1 = __builtin_amdgcn_wmma_f32_16x16x32_f16(false, a, false, b1v, (short)0, c1, false, false);
  }

  // ---- bias + relu + transpose via LDS into A-operand layout ----
  int col = l15;
  float bb0 = bias0[col];
  float bb1 = bias0[col + 16];
  _Float16* h = &hid[wave][0];
#pragma unroll
  for (int r = 0; r < 8; ++r) {
    int m = r + 8 * hl;                  // C/D layout: lane holds col, rows r+8*half
    h[m * 32 + col]      = (_Float16)fmaxf(c0[r] + bb0, 0.0f);
    h[m * 32 + col + 16] = (_Float16)fmaxf(c1[r] + bb1, 0.0f);
  }
  // Same-wave DS ordering guarantees store->load visibility (ISA ch.11).

  // ---- GEMM 2: [16 x 32] @ [32 x 32], B from LDS ----
  v8h a2lo = *(const v8h*)(h + l15 * 32 + 8 * hl);
  v8h a2hi = *(const v8h*)(h + l15 * 32 + 16 + 8 * hl);
  v16h a2 = __builtin_shufflevector(a2lo, a2hi, 0,1,2,3,4,5,6,7,8,9,10,11,12,13,14,15);
  v16h w20 = *(const v16h*)(&wlds[MAXCHUNKS * 1024 + ((0 * 32) + lane) * 16]);
  v16h w21 = *(const v16h*)(&wlds[MAXCHUNKS * 1024 + ((1 * 32) + lane) * 16]);
  v8f d0 = {}; v8f d1 = {};
  d0 = __builtin_amdgcn_wmma_f32_16x16x32_f16(false, a2, false, w20, (short)0, d0, false, false);
  d1 = __builtin_amdgcn_wmma_f32_16x16x32_f16(false, a2, false, w21, (short)0, d1, false, false);

  float cb0 = bias1[col];
  float cb1 = bias1[col + 16];
#pragma unroll
  for (int r = 0; r < 8; ++r) {
    int m = r + 8 * hl;
    int e = tileBase + m;
    float v0 = d0[r] + cb0;
    float v1 = d1[r] + cb1;
    if (doScatter) {                     // seg-sum of PRE-relu values
      int tgt = rowI[e];
      atomicAdd(&sAcc[(size_t)tgt * 32 + col],      v0);
      atomicAdd(&sAcc[(size_t)tgt * 32 + col + 16], v1);
    }
    Eout[(size_t)e * 32 + col]      = (_Float16)fmaxf(v0, 0.0f);
    Eout[(size_t)e * 32 + col + 16] = (_Float16)fmaxf(v1, 0.0f);
  }
}

// ---------------------------------------------------------------------------
// Small utility kernels
// ---------------------------------------------------------------------------
__global__ void zero_f32_kernel(float* p, int n) {
  int i = blockIdx.x * blockDim.x + threadIdx.x;
  if (i < n) p[i] = 0.0f;
}
__global__ void count_edges_kernel(const int* __restrict__ rowI, float* cnt, int nE) {
  int i = blockIdx.x * blockDim.x + threadIdx.x;
  if (i < nE) atomicAdd(&cnt[rowI[i]], 1.0f);
}
__global__ void inv_count_kernel(float* cnt, int n) {
  int i = blockIdx.x * blockDim.x + threadIdx.x;
  if (i < n) cnt[i] = 1.0f / fmaxf(cnt[i], 1.0f);
}
__global__ void finalize_x_kernel(const float* __restrict__ s, const float* __restrict__ invc,
                                  float* __restrict__ x, int n) {
  int i = blockIdx.x * blockDim.x + threadIdx.x;
  if (i < n) x[i] = fmaxf(s[i] * invc[i >> 5], 0.0f);
}
__global__ void copy_beta_kernel(const float* __restrict__ beta, float* __restrict__ out, int n) {
  int i = blockIdx.x * blockDim.x + threadIdx.x;
  if (i < n) out[i] = beta[i];
}

// ---------------------------------------------------------------------------
// Heads: out01 = e4@wl01+b, out02 = e4@wl02+b,
//        edge_x = normalize(out01@wl1+bl1+edge_attr), probs = sigmoid(out02@wl2+bl2)
// One thread per edge; weights staged in LDS.
// ---------------------------------------------------------------------------
__global__ __launch_bounds__(256)
void heads_kernel(const _Float16* __restrict__ e4, const float* __restrict__ ea,
                  const float* __restrict__ wl01, const float* __restrict__ bl01,
                  const float* __restrict__ wl02, const float* __restrict__ bl02,
                  const float* __restrict__ wl1,  const float* __restrict__ bl1,
                  const float* __restrict__ wl2,  const float* __restrict__ bl2,
                  float* __restrict__ probs, float* __restrict__ exOut, int nE) {
  __shared__ float sw01[1024], sw02[1024], sw1[128], sw2[32];
  __shared__ float sb01[32], sb02[32], sb1[4], sb2;
  for (int i = threadIdx.x; i < 1024; i += 256) { sw01[i] = wl01[i]; sw02[i] = wl02[i]; }
  for (int i = threadIdx.x; i < 128; i += 256) sw1[i] = wl1[i];
  if (threadIdx.x < 32) { sw2[threadIdx.x] = wl2[threadIdx.x];
                          sb01[threadIdx.x] = bl01[threadIdx.x];
                          sb02[threadIdx.x] = bl02[threadIdx.x]; }
  if (threadIdx.x < 4) sb1[threadIdx.x] = bl1[threadIdx.x];
  if (threadIdx.x == 0) sb2 = bl2[0];
  __syncthreads();

  int e = blockIdx.x * 256 + threadIdx.x;
  if (e >= nE) return;

  float r[32];
#pragma unroll
  for (int i = 0; i < 32; ++i) r[i] = (float)e4[(size_t)e * 32 + i];

  float o1[32], o2[32];
#pragma unroll 4
  for (int j = 0; j < 32; ++j) {
    float a1 = sb01[j], a2 = sb02[j];
#pragma unroll
    for (int i = 0; i < 32; ++i) {
      a1 = fmaf(r[i], sw01[i * 32 + j], a1);
      a2 = fmaf(r[i], sw02[i * 32 + j], a2);
    }
    o1[j] = a1; o2[j] = a2;
  }

  float ex[4];
#pragma unroll
  for (int j = 0; j < 4; ++j) {
    float a = sb1[j] + ea[(size_t)e * 4 + j];
#pragma unroll
    for (int i = 0; i < 32; ++i) a = fmaf(o1[i], sw1[i * 4 + j], a);
    ex[j] = a;
  }
  float nrm = sqrtf(ex[0]*ex[0] + ex[1]*ex[1] + ex[2]*ex[2] + ex[3]*ex[3]);
  float inv = 1.0f / fmaxf(nrm, 1e-12f);
#pragma unroll
  for (int j = 0; j < 4; ++j) exOut[(size_t)e * 4 + j] = ex[j] * inv;

  float p = sb2;
#pragma unroll
  for (int i = 0; i < 32; ++i) p = fmaf(o2[i], sw2[i], p);
  probs[e] = 1.0f / (1.0f + expf(-p));
}

// ---------------------------------------------------------------------------
// Orchestration
// ---------------------------------------------------------------------------
static inline int cdiv(long a, int b) { return (int)((a + b - 1) / b); }

extern "C" void kernel_launch(void* const* d_in, const int* in_sizes, int n_in,
                              void* d_out, int out_size, void* d_ws, size_t ws_size,
                              hipStream_t stream) {
  const int N = NNODES, E = NEDGES;
  (void)in_sizes; (void)n_in; (void)out_size; (void)ws_size;

  const int*   eidx = (const int*)d_in[1];
  const int*   rowI = eidx;
  const int*   colI = eidx + E;
  const float* ea   = (const float*)d_in[2];
  const float* beta = (const float*)d_in[3];
  const float* W[12]; const float* B[12];
  for (int i = 0; i < 12; ++i) { W[i] = (const float*)d_in[4 + 2*i]; B[i] = (const float*)d_in[5 + 2*i]; }
  // W: 0=w10 1=w11 2=w20 3=w21 4=w30 5=w31 6=w40 7=w41 8=wl01 9=wl02 10=wl1 11=wl2

  // ---- workspace carve (all 256B aligned) ----
  char* w = (char*)d_ws;
  auto carve = [&](size_t bytes) { void* p = (void*)w; w += (bytes + 255) & ~(size_t)255; return p; };
  _Float16* e1  = (_Float16*)carve((size_t)E * 32 * 2);
  _Float16* e2  = (_Float16*)carve((size_t)E * 32 * 2);
  _Float16* e3  = (_Float16*)carve((size_t)E * 32 * 2);
  _Float16* e4  = (_Float16*)carve((size_t)E * 32 * 2);
  float*    x1  = (float*)carve((size_t)N * 32 * 4);
  float*    x2  = (float*)carve((size_t)N * 32 * 4);
  float*    x3  = (float*)carve((size_t)N * 32 * 4);
  float*    sAc = (float*)carve((size_t)N * 32 * 4);
  float*    inv = (float*)carve((size_t)N * 4);
  _Float16* Ain = (_Float16*)carve((size_t)E * 192 * 2);
  _Float16* wp[8];
  for (int i = 0; i < 8; ++i) wp[i] = (_Float16*)carve(16384);

  // ---- pack weights into B-operand layout ----
  const int fins[8]   = {4, 32, 100, 32, 192, 32, 192, 32};
  const int chunks[8] = {1, 1, 4, 1, 6, 1, 6, 1};
  for (int i = 0; i < 8; ++i)
    pack_w_kernel<<<chunks[i] * 2, 32, 0, stream>>>(W[i], wp[i], fins[i]);

  // ---- segment counts (row index is layer-invariant) ----
  zero_f32_kernel<<<cdiv(N, 256), 256, 0, stream>>>(inv, N);
  count_edges_kernel<<<cdiv(E, 256), 256, 0, stream>>>(rowI, inv, E);
  inv_count_kernel<<<cdiv(N, 256), 256, 0, stream>>>(inv, N);

  const int gemmBlocks = E / 16 / 8;   // 3125, exact

  // ---- conv1 ----
  pack_a_kernel<<<cdiv((long)E * 32, 256), 256, 0, stream>>>(Ain, 32, 1, rowI, colI, ea,
                                                             nullptr, nullptr, nullptr, nullptr, E);
  zero_f32_kernel<<<cdiv((long)N * 32, 256), 256, 0, stream>>>(sAc, N * 32);
  edge_mlp_wmma_kernel<<<gemmBlocks, 256, 0, stream>>>(Ain, 1, 32, wp[0], B[0], wp[1], B[1],
                                                       e1, rowI, sAc, 1, E);
  finalize_x_kernel<<<cdiv((long)N * 32, 256), 256, 0, stream>>>(sAc, inv, x1, N * 32);

  // ---- conv2 ----
  pack_a_kernel<<<cdiv((long)E * 128, 256), 256, 0, stream>>>(Ain, 128, 2, rowI, colI, ea,
                                                              x1, nullptr, e1, nullptr, E);
  zero_f32_kernel<<<cdiv((long)N * 32, 256), 256, 0, stream>>>(sAc, N * 32);
  edge_mlp_wmma_kernel<<<gemmBlocks, 256, 0, stream>>>(Ain, 4, 128, wp[2], B[2], wp[3], B[3],
                                                       e2, rowI, sAc, 1, E);
  finalize_x_kernel<<<cdiv((long)N * 32, 256), 256, 0, stream>>>(sAc, inv, x2, N * 32);

  // ---- conv3 ----
  pack_a_kernel<<<cdiv((long)E * 192, 256), 256, 0, stream>>>(Ain, 192, 3, rowI, colI, ea,
                                                              x2, x1, e2, e1, E);
  zero_f32_kernel<<<cdiv((long)N * 32, 256), 256, 0, stream>>>(sAc, N * 32);
  edge_mlp_wmma_kernel<<<gemmBlocks, 256, 0, stream>>>(Ain, 6, 192, wp[4], B[4], wp[5], B[5],
                                                       e3, rowI, sAc, 1, E);
  finalize_x_kernel<<<cdiv((long)N * 32, 256), 256, 0, stream>>>(sAc, inv, x3, N * 32);

  // ---- conv4 (node aggregate unused downstream -> no scatter) ----
  pack_a_kernel<<<cdiv((long)E * 192, 256), 256, 0, stream>>>(Ain, 192, 3, rowI, colI, ea,
                                                              x3, x2, e3, e2, E);
  edge_mlp_wmma_kernel<<<gemmBlocks, 256, 0, stream>>>(Ain, 6, 192, wp[6], B[6], wp[7], B[7],
                                                       e4, rowI, nullptr, 0, E);

  // ---- heads + beta passthrough ----
  float* out = (float*)d_out;
  heads_kernel<<<cdiv(E, 256), 256, 0, stream>>>(e4, ea, W[8], B[8], W[9], B[9],
                                                 W[10], B[10], W[11], B[11],
                                                 out, out + E, E);
  copy_beta_kernel<<<cdiv((long)N * 3, 256), 256, 0, stream>>>(beta, out + (size_t)5 * E, N * 3);
}
